// EdgeDecoder_83820581748943
// MI455X (gfx1250) — matrix operand order repaired and checked by reading, not compiled, not run on hardware
//
#include <hip/hip_runtime.h>

// EdgeDecoder for MI455X (gfx1250, wave32, WMMA).
//
// score[e] = w2 . relu( concat(z[src[e]], z[dst[e]]) @ w1^T + b1 ) + b2
//
// Compute-bound (~33 GFLOP; z=51MB and w1=128KB are L2-resident, HBM ~60MB):
// layer-1 GEMM on v_wmma_f32_16x16x32_f16 (f16 in, f32 accum). w1 staged once
// per block into LDS as f16 (vectorized f32x8 -> cvt_pk -> ds_store_b128,
// padded 528B rows for bank-conflict-free B-fragment loads) and amortized over
// 512 edges/block (4 edge-tiles per wave). Layer 2 (128->1) fused in
// registers + wave shuffles.

typedef __attribute__((ext_vector_type(16))) _Float16 v16h;
typedef __attribute__((ext_vector_type(8)))  float    v8f;

#define HIDDEN     128
#define K2         256           // 2*HIDDEN
#define LDS_STRIDE 264           // halfs per padded w1 row (528 B = 33*16 B)
#define EDGES_PER_WAVE_TILE 16
#define TILES_PER_WAVE      4
#define WAVES_PER_BLOCK     8
#define EDGES_PER_BLOCK (EDGES_PER_WAVE_TILE * TILES_PER_WAVE * WAVES_PER_BLOCK) // 512

union BFrag32 { uint4 q[2]; v16h v; };
union Pack8   { _Float16 h[8]; uint4 q; };

extern "C" __global__ __launch_bounds__(256)
void edge_decoder_wmma(const float* __restrict__ z,          // [N_NODES, 128]
                       const int*   __restrict__ edge_index, // [2, E] (int32)
                       const float* __restrict__ w1,         // [128, 256] row-major
                       const float* __restrict__ b1,         // [128]
                       const float* __restrict__ w2,         // [128] (w2 row 0)
                       const float* __restrict__ b2,         // [1]
                       float*       __restrict__ out,        // [E]
                       int E)
{
    extern __shared__ char smem_raw[];
    _Float16* w1h = (_Float16*)smem_raw;                       // 128 x 264 halfs
    float*    b1s = (float*)(smem_raw + HIDDEN * LDS_STRIDE * 2);
    float*    w2s = b1s + HIDDEN;

    const int tid = threadIdx.x;

    // ---- Stage w1 (f32 -> f16) into LDS, 8 elements per chunk ----
    // 32768 elems / 8 = 4096 chunks; 16 chunks per thread; consecutive tids
    // handle consecutive chunks (coalesced 32B global reads, aligned 16B LDS
    // stores). 8-half chunks never straddle a 256-elem row.
    #pragma unroll
    for (int c = 0; c < 16; ++c) {
        const int ch   = c * 256 + tid;       // chunk index 0..4095
        const int row  = ch >> 5;             // 32 chunks per row
        const int colh = (ch & 31) * 8;       // half offset within row
        const float* p = w1 + ch * 8;
        float4 r0 = *(const float4*)(p);
        float4 r1 = *(const float4*)(p + 4);
        Pack8 pk;
        pk.h[0] = (_Float16)r0.x; pk.h[1] = (_Float16)r0.y;
        pk.h[2] = (_Float16)r0.z; pk.h[3] = (_Float16)r0.w;
        pk.h[4] = (_Float16)r1.x; pk.h[5] = (_Float16)r1.y;
        pk.h[6] = (_Float16)r1.z; pk.h[7] = (_Float16)r1.w;
        *(uint4*)(w1h + row * LDS_STRIDE + colh) = pk.q;
    }
    if (tid < HIDDEN) { b1s[tid] = b1[tid]; w2s[tid] = w2[tid]; }
    __syncthreads();

    const int lane   = tid & 31;
    const int wave   = tid >> 5;
    const int laneLo = lane & 15;     // column/row position within half-group
    const int hiGrp  = lane >> 4;     // 0 or 1
    const int kbaseA = hiGrp * 8;     // A-fragment K base (16-bit A 16x32 layout)
    const int kbaseB = hiGrp * 8;     // B-fragment K base (matches A-fragment K order)

    const float bias2 = b2[0];
    const int   waveBase = blockIdx.x * EDGES_PER_BLOCK
                         + wave * (EDGES_PER_WAVE_TILE * TILES_PER_WAVE);

    // ---- 4 edge tiles per wave against the same LDS-resident w1 ----
    for (int t = 0; t < TILES_PER_WAVE; ++t) {
        const int etile = waveBase + t * EDGES_PER_WAVE_TILE;
        if (etile >= E) break;                // uniform per wave: EXEC stays full

        // A-matrix row for this lane = edge (etile + laneLo); clamp for the
        // partial tail tile (branchless; stores guarded below).
        int eA = etile + laneLo;
        if (eA > E - 1) eA = E - 1;
        const int srcn = edge_index[eA];
        const int dstn = edge_index[E + eA];
        const float* __restrict__ zsrc = z + (long)srcn * HIDDEN;
        const float* __restrict__ zdst = z + (long)dstn * HIDDEN;

        // ---- Preload all 8 A fragments (K = 256 = 8 x 32) ----
        // kt 0..3 -> z[src] (K 0..127), kt 4..7 -> z[dst] (concat is free).
        // ISA A-16x32 layout: halves 0..7 -> K = kt*32 + kbaseA + t,
        //                     halves 8..15 -> K = kt*32 + 16 + kbaseA + (t-8).
        v16h a[8];
        #pragma unroll
        for (int kt = 0; kt < 8; ++kt) {
            const float* base = (kt < 4) ? zsrc : zdst;
            const float* p = base + (kt & 3) * 32 + kbaseA;
            float4 r0 = *(const float4*)(p);
            float4 r1 = *(const float4*)(p + 4);
            float4 r2 = *(const float4*)(p + 16);
            float4 r3 = *(const float4*)(p + 20);
            v16h f;
            f[0]  = (_Float16)r0.x; f[1]  = (_Float16)r0.y;
            f[2]  = (_Float16)r0.z; f[3]  = (_Float16)r0.w;
            f[4]  = (_Float16)r1.x; f[5]  = (_Float16)r1.y;
            f[6]  = (_Float16)r1.z; f[7]  = (_Float16)r1.w;
            f[8]  = (_Float16)r2.x; f[9]  = (_Float16)r2.y;
            f[10] = (_Float16)r2.z; f[11] = (_Float16)r2.w;
            f[12] = (_Float16)r3.x; f[13] = (_Float16)r3.y;
            f[14] = (_Float16)r3.z; f[15] = (_Float16)r3.w;
            a[kt] = f;
        }

        // Per-lane partial scores; s[r] belongs to edge (etile + 8*hiGrp + r)
        // per the 16x16 f32 D layout (VGPR r: lanes 0-15 -> M=r, 16-31 -> M=r+8).
        float s[8] = {0.f, 0.f, 0.f, 0.f, 0.f, 0.f, 0.f, 0.f};

        // ---- Layer 1 GEMM via WMMA: 8 N-tiles x 8 K-steps ----
        #pragma unroll
        for (int jt = 0; jt < 8; ++jt) {
            const int j = jt * 16 + laneLo;   // output feature this lane owns
            v8f acc = {};
            const _Float16* __restrict__ wrow = w1h + j * LDS_STRIDE + kbaseB;
            #pragma unroll
            for (int kt = 0; kt < 8; ++kt) {
                // B fragment (32x16): lane col = laneLo, halves 0..7 -> K=kbaseB+t, 8..15 -> K=16+kbaseB+(t-8).
                // 528B row stride => 16 lanes hit 64 distinct LDS banks.
                const _Float16* bp = wrow + kt * 32;
                BFrag32 u;
                u.q[0] = *(const uint4*)(bp);
                u.q[1] = *(const uint4*)(bp + 16);
                acc = __builtin_amdgcn_wmma_f32_16x16x32_f16(
                    false, a[kt], false, u.v, (short)0, acc, false, false);
            }
            // ---- Fuse bias + ReLU + layer 2 (dot with w2) ----
            const float bj = b1s[j];
            const float wj = w2s[j];
            #pragma unroll
            for (int r = 0; r < 8; ++r) {
                float h = acc[r] + bj;
                h = (h > 0.f) ? h : 0.f;
                s[r] += h * wj;
            }
        }

        // ---- Reduce the 16 feature-columns (lanes within each half-group) ----
        #pragma unroll
        for (int r = 0; r < 8; ++r) {
            float v = s[r];
            v += __shfl_xor(v, 8, 16);
            v += __shfl_xor(v, 4, 16);
            v += __shfl_xor(v, 2, 16);
            v += __shfl_xor(v, 1, 16);
            s[r] = v;
        }

        if (laneLo == 0) {
            #pragma unroll
            for (int r = 0; r < 8; ++r) {
                const int e = etile + hiGrp * 8 + r;
                if (e < E) out[e] = s[r] + bias2;
            }
        }
    }
}

extern "C" void kernel_launch(void* const* d_in, const int* in_sizes, int n_in,
                              void* d_out, int out_size, void* d_ws, size_t ws_size,
                              hipStream_t stream) {
    const float* z          = (const float*)d_in[0];
    const int*   edge_index = (const int*)  d_in[1];   // [2, E]
    const float* w1         = (const float*)d_in[2];   // [128, 256]
    const float* b1         = (const float*)d_in[3];   // [128]
    const float* w2         = (const float*)d_in[4];   // [1, 128]
    const float* b2         = (const float*)d_in[5];   // [1]
    float* out = (float*)d_out;

    const int E = out_size;                            // score is [E, 1]
    const int blocks = (E + EDGES_PER_BLOCK - 1) / EDGES_PER_BLOCK;
    const size_t smem = (size_t)HIDDEN * LDS_STRIDE * sizeof(_Float16)  // w1 f16 padded
                      + 2 * HIDDEN * sizeof(float);                     // b1, w2

    edge_decoder_wmma<<<dim3(blocks), dim3(256), smem, stream>>>(
        z, edge_index, w1, b1, w2, b2, out, E);
}
